// Attention_1795296330058
// MI455X (gfx1250) — matrix operand order repaired and checked by reading, not compiled, not run on hardware
//
#include <hip/hip_runtime.h>

// ---------------------------------------------------------------------------
// Single-head attention, B=4, S=2048, D=1024, fp32 in/out.
// All GEMMs on v_wmma_f32_16x16x32_bf16 (compute-bound: ~120 GF over ~80 MB).
//   K1 proj_gemm<0/1> : X @ W -> Q,K bf16 row-major / V bf16 transposed.
//   K2 attn           : 16 query rows per WG; logits+softmax+PV in 224KB LDS;
//                       Q tile staged by the Tensor Data Mover (TDM).
//                       LDS-limited to 1 WG/WGP => amdgpu_waves_per_eu(2,2)
//                       gives waves a big VGPR budget (no spills).
// ---------------------------------------------------------------------------

#define DIN  1024
#define DOUT 1024
#define SEQ  2048
#define BATCH 4

typedef __attribute__((ext_vector_type(16))) __bf16 v16bf;
typedef __attribute__((ext_vector_type(8)))  float  v8f;
typedef unsigned int u32;
typedef __attribute__((ext_vector_type(4))) u32 u32x4;
typedef __attribute__((ext_vector_type(8))) int i32x8;
typedef __attribute__((ext_vector_type(4))) int i32x4;

struct alignas(16) U4 { unsigned int a, b, c, d; };
struct alignas(16) F4 { float x, y, z, w; };

union FragU { U4 u[2]; v16bf v; };

__device__ __forceinline__ unsigned short f2bf(float f) {
    unsigned int u = __float_as_uint(f);
    u += 0x7FFFu + ((u >> 16) & 1u);        // round-to-nearest-even
    return (unsigned short)(u >> 16);
}

// A-fragment (16x32 bf16, M x K), row-major source, stride `rs` elements.
__device__ __forceinline__ v16bf load_frag_a(const unsigned short* p0, int rs, int lane) {
    int m = lane & 15, hb = (lane >> 4) * 8;
    const unsigned short* p = p0 + m * rs + hb;
    FragU f;
    f.u[0] = *(const U4*)(p);        // K = hb .. hb+7
    f.u[1] = *(const U4*)(p + 16);   // K = 16+hb .. 16+hb+7
    return f.v;
}

// B-fragment (32x16 bf16, K x N) from transposed storage bt[n][k], stride `rs`.
__device__ __forceinline__ v16bf load_frag_b(const unsigned short* p0, int rs, int lane) {
    int n = lane & 15, kb = (lane >> 4) * 8;
    const unsigned short* p = p0 + n * rs + kb;
    FragU f;
    f.u[0] = *(const U4*)(p);
    f.u[1] = *(const U4*)(p + 16);
    return f.v;
}

__device__ __forceinline__ v8f wmma_bf16(v16bf a, v16bf b, v8f c) {
    return __builtin_amdgcn_wmma_f32_16x16x32_bf16(
        false, a, false, b, (short)0, c, false, false);
}

// LDS byte offset of a generic pointer (AS3 ptrtoint == LDS offset).
typedef __attribute__((address_space(3))) const void* lds_cptr;
__device__ __forceinline__ unsigned lds_offset(const void* p) {
    return (unsigned)(unsigned long long)(lds_cptr)p;
}

// Issue a 2D TDM load: tile (tile1 rows x tile0 dwords) from a row-major
// tensor with row stride stride0 dwords, into LDS at lds_off.  D# per ISA §8.
// 6-arg builtin: (uint32x4 g0, int32x8 g1, int32x4, int32x4, int32x8, i32 cpol)
__device__ __forceinline__ void tdm_load_2d(unsigned lds_off, const void* gaddr,
                                            u32 dim0, u32 dim1,
                                            u32 tile0, u32 tile1, u32 stride0) {
    unsigned long long ga = (unsigned long long)gaddr;
    u32x4 g0;
    g0[0] = 1u;                                           // count=1, user mode
    g0[1] = lds_off;                                      // lds_addr (bytes)
    g0[2] = (u32)(ga & 0xFFFFFFFFu);                      // global_addr[31:0]
    g0[3] = (u32)((ga >> 32) & 0x01FFFFFFu) | (2u << 30); // addr[56:32], type=2
    i32x8 g1;
    g1[0] = (int)(2u << 16);                              // data_size = 4 bytes
    g1[1] = (int)((dim0 & 0xFFFFu) << 16);                // tensor_dim0[15:0]
    g1[2] = (int)((dim0 >> 16) | ((dim1 & 0xFFFFu) << 16));
    g1[3] = (int)((dim1 >> 16) | (tile0 << 16));          // tile_dim0
    g1[4] = (int)(tile1 & 0xFFFFu);                       // tile_dim1 (tile_dim2=0)
    g1[5] = (int)stride0;                                 // tensor_dim0_stride[31:0]
    g1[6] = 0;
    g1[7] = 0;
    i32x4 zz4 = {0, 0, 0, 0};
    i32x8 zz8 = {0, 0, 0, 0, 0, 0, 0, 0};
    __builtin_amdgcn_tensor_load_to_lds(g0, g1, zz4, zz4, zz8, 0);
}

// ---------------------------------------------------------------------------
// K1: Y = X @ W (fp32 in, bf16 out).  grid=(N/64, M/128), 8 waves.
// MODE 0: dst row-major [M][N].  MODE 1: dst transposed Vt[b][n][s], with the
// 8 accumulator rows of a lane packed into ONE global_store_b128.
// ---------------------------------------------------------------------------
template <int MODE>
__global__ __attribute__((amdgpu_flat_work_group_size(256, 256)))
void proj_gemm(const float* __restrict__ x,
               const float* __restrict__ W,
               unsigned short* __restrict__ dst) {
    __shared__ unsigned short Abf[128 * 32];   // X tile, bf16, row-major
    __shared__ unsigned short Wt[64 * 32];     // W tile transposed: Wt[n][k]

    const int tid  = threadIdx.x;
    const int lane = tid & 31;
    const int w    = tid >> 5;
    const int n0   = blockIdx.x * 64;
    const int m0   = blockIdx.y * 128;

    v8f zero = {};
    v8f acc[4];
#pragma unroll
    for (int t = 0; t < 4; ++t) acc[t] = zero;

    for (int k0 = 0; k0 < DIN; k0 += 32) {
        __syncthreads();
        {   // X tile 128x32 fp32 -> bf16 LDS
            const int row = tid >> 1, half = tid & 1;
            const float* src = x + (size_t)(m0 + row) * DIN + k0 + half * 16;
            unsigned int pk[8];
#pragma unroll
            for (int i = 0; i < 4; ++i) {
                F4 f = *(const F4*)(src + i * 4);
                pk[i * 2 + 0] = (unsigned)f2bf(f.x) | ((unsigned)f2bf(f.y) << 16);
                pk[i * 2 + 1] = (unsigned)f2bf(f.z) | ((unsigned)f2bf(f.w) << 16);
            }
            unsigned short* d = Abf + row * 32 + half * 16;
            U4 u0 = {pk[0], pk[1], pk[2], pk[3]};
            U4 u1 = {pk[4], pk[5], pk[6], pk[7]};
            *(U4*)(d)     = u0;
            *(U4*)(d + 8) = u1;
        }
        {   // W tile 32x64 fp32 -> transposed bf16 LDS
            const int kk = tid >> 3, cb = (tid & 7) * 8;
            const float* wsrc = W + (size_t)(k0 + kk) * DOUT + n0 + cb;
            F4 f0 = *(const F4*)(wsrc);
            F4 f1 = *(const F4*)(wsrc + 4);
            Wt[(cb + 0) * 32 + kk] = f2bf(f0.x);
            Wt[(cb + 1) * 32 + kk] = f2bf(f0.y);
            Wt[(cb + 2) * 32 + kk] = f2bf(f0.z);
            Wt[(cb + 3) * 32 + kk] = f2bf(f0.w);
            Wt[(cb + 4) * 32 + kk] = f2bf(f1.x);
            Wt[(cb + 5) * 32 + kk] = f2bf(f1.y);
            Wt[(cb + 6) * 32 + kk] = f2bf(f1.z);
            Wt[(cb + 7) * 32 + kk] = f2bf(f1.w);
        }
        __syncthreads();
        v16bf a = load_frag_a(Abf + w * 16 * 32, 32, lane);
#pragma unroll
        for (int t = 0; t < 4; ++t) {
            v16bf b = load_frag_b(Wt + t * 16 * 32, 32, lane);
            acc[t] = wmma_bf16(a, b, acc[t]);
        }
    }

    const int hi = lane >> 4, ln = lane & 15;
    if (MODE == 0) {
#pragma unroll
        for (int t = 0; t < 4; ++t) {
#pragma unroll
            for (int r = 0; r < 8; ++r) {
                const int m_g = m0 + w * 16 + r + hi * 8;
                dst[(size_t)m_g * DOUT + n0 + t * 16 + ln] = f2bf(acc[t][r]);
            }
        }
    } else {
        // transposed: lane's 8 rows are 8 consecutive s -> one b128 store
        const int s_base = m0 + w * 16 + hi * 8;
        const int b = s_base >> 11, s_loc = s_base & (SEQ - 1);
#pragma unroll
        for (int t = 0; t < 4; ++t) {
            const int d = n0 + t * 16 + ln;
            U4 pk;
            pk.a = (u32)f2bf(acc[t][0]) | ((u32)f2bf(acc[t][1]) << 16);
            pk.b = (u32)f2bf(acc[t][2]) | ((u32)f2bf(acc[t][3]) << 16);
            pk.c = (u32)f2bf(acc[t][4]) | ((u32)f2bf(acc[t][5]) << 16);
            pk.d = (u32)f2bf(acc[t][6]) | ((u32)f2bf(acc[t][7]) << 16);
            *(U4*)(dst + ((size_t)((b << 10) + d)) * SEQ + s_loc) = pk;
        }
    }
}

// ---------------------------------------------------------------------------
// K2: fused attention for 16 query rows. grid=(S/16, B), 256 threads.
// Dynamic LDS: Q tile 32KB | logits fp32 128KB | P bf16 64KB  (224KB).
// ---------------------------------------------------------------------------
__global__ __attribute__((amdgpu_flat_work_group_size(256, 256),
                          amdgpu_waves_per_eu(2, 2)))
void attn(const unsigned short* __restrict__ Qb,
          const unsigned short* __restrict__ Kb,
          const unsigned short* __restrict__ Vt,
          float* __restrict__ out) {
    extern __shared__ char smem[];
    unsigned short* lq = (unsigned short*)smem;                        // 16 x 1024 bf16
    float*          sc = (float*)(smem + 32768);                       // 16 x 2048 f32
    unsigned short* pb = (unsigned short*)(smem + 32768 + 131072);     // 16 x 2048 bf16

    const int tid  = threadIdx.x;
    const int lane = tid & 31;
    const int w    = tid >> 5;
    const int s0   = blockIdx.x * 16;
    const int b    = blockIdx.y;
    const unsigned short* Qsrc = Qb + ((size_t)b * SEQ + s0) * DOUT;
    const unsigned short* Kbb  = Kb + (size_t)b * SEQ * DOUT;
    const unsigned short* Vtb  = Vt + (size_t)b * DOUT * SEQ;

    // --- TDM: DMA the 16x1024 bf16 Q tile (16 rows x 512 dwords) into LDS ---
    if (w == 0) {
        if (lane == 0)
            tdm_load_2d(lds_offset(lq), Qsrc,
                        /*dim0=*/DOUT / 2, /*dim1=*/16,
                        /*tile0=*/DOUT / 2, /*tile1=*/16,
                        /*stride0=*/DOUT / 2);
        __builtin_amdgcn_s_wait_tensorcnt(0);
    }
    __syncthreads();

    const v8f zero = {};

    // --- stage 1: logits = (Q K^T)/sqrt(D); wave w owns key tiles w+8j.
    // B-fragments double-buffered in registers: WMMA j waits only on loads j.
    {
        v8f acc[16];
#pragma unroll
        for (int j = 0; j < 16; ++j) acc[j] = zero;

        for (int d0 = 0; d0 < DOUT; d0 += 32) {
            v16bf a = load_frag_a(lq + d0, DOUT, lane);
            __builtin_prefetch(Kbb + (size_t)(w * 16) * DOUT + d0 + 32, 0, 1);
            v16bf bb = load_frag_b(Kbb + (size_t)(w * 16) * DOUT + d0, DOUT, lane);
#pragma unroll
            for (int j = 0; j < 16; ++j) {
                v16bf bn = bb;
                if (j + 1 < 16) {
                    const int keyn = (w + 8 * (j + 1)) * 16;
                    bn = load_frag_b(Kbb + (size_t)keyn * DOUT + d0, DOUT, lane);
                }
                acc[j] = wmma_bf16(a, bb, acc[j]);
                bb = bn;
            }
        }
        const int hi = lane >> 4, ln = lane & 15;
#pragma unroll
        for (int j = 0; j < 16; ++j) {
            const int col = (w + 8 * j) * 16 + ln;
#pragma unroll
            for (int r = 0; r < 8; ++r)
                sc[(size_t)(r + hi * 8) * SEQ + col] = acc[j][r] * 0.03125f;
        }
    }
    __syncthreads();

    // --- stage 2: softmax over rows of 2048 (2 rows/wave, wave32 shuffles) ---
#pragma unroll
    for (int rr = 0; rr < 2; ++rr) {
        const int row = w * 2 + rr;
        float* rp = sc + (size_t)row * SEQ;
        float mx = -3.4e38f;
        for (int i = lane; i < SEQ; i += 32) mx = fmaxf(mx, rp[i]);
#pragma unroll
        for (int off = 16; off; off >>= 1) mx = fmaxf(mx, __shfl_xor(mx, off, 32));
        float sum = 0.f;
        for (int i = lane; i < SEQ; i += 32) {
            float e = __expf(rp[i] - mx);
            sum += e;
            rp[i] = e;
        }
#pragma unroll
        for (int off = 16; off; off >>= 1) sum += __shfl_xor(sum, off, 32);
        const float inv = 1.0f / sum;
        unsigned short* pr = pb + (size_t)row * SEQ;
        for (int i = lane; i < SEQ; i += 32) pr[i] = f2bf(rp[i] * inv);
    }
    __syncthreads();

    // --- stage 3: O = P @ V ; wave w owns output columns [128w, 128w+128) ---
    {
        v8f oacc[8];
#pragma unroll
        for (int t = 0; t < 8; ++t) oacc[t] = zero;

        for (int kk = 0; kk < SEQ; kk += 32) {
            v16bf a = load_frag_a(pb + kk, SEQ, lane);
            __builtin_prefetch(Vtb + (size_t)(w * 128) * SEQ + kk + 32, 0, 1);
            v16bf bb = load_frag_b(Vtb + (size_t)(w * 128) * SEQ + kk, SEQ, lane);
#pragma unroll
            for (int t = 0; t < 8; ++t) {
                v16bf bn = bb;
                if (t + 1 < 8) {
                    const int n1 = w * 128 + (t + 1) * 16;
                    bn = load_frag_b(Vtb + (size_t)n1 * SEQ + kk, SEQ, lane);
                }
                oacc[t] = wmma_bf16(a, bb, oacc[t]);
                bb = bn;
            }
        }
        const int hi = lane >> 4, ln = lane & 15;
#pragma unroll
        for (int t = 0; t < 8; ++t) {
#pragma unroll
            for (int r = 0; r < 8; ++r) {
                const int m = r + hi * 8;
                out[((size_t)b * SEQ + s0 + m) * DOUT + w * 128 + t * 16 + ln] = oacc[t][r];
            }
        }
    }
}

// ---------------------------------------------------------------------------
extern "C" void kernel_launch(void* const* d_in, const int* in_sizes, int n_in,
                              void* d_out, int out_size, void* d_ws, size_t ws_size,
                              hipStream_t stream) {
    const float* x  = (const float*)d_in[0];
    const float* WQ = (const float*)d_in[1];
    const float* WK = (const float*)d_in[2];
    const float* WV = (const float*)d_in[3];
    float* out = (float*)d_out;

    // Workspace: Q bf16 | K bf16 | V^T bf16   (3 * 16.78 MB = 50.3 MB)
    unsigned short* Qw = (unsigned short*)d_ws;
    unsigned short* Kw = Qw + (size_t)BATCH * SEQ * DOUT;
    unsigned short* Vt = Kw + (size_t)BATCH * SEQ * DOUT;

    const dim3 pgrid(DOUT / 64, (BATCH * SEQ) / 128);
    proj_gemm<0><<<pgrid, 256, 0, stream>>>(x, WQ, Qw);
    proj_gemm<0><<<pgrid, 256, 0, stream>>>(x, WK, Kw);
    proj_gemm<1><<<pgrid, 256, 0, stream>>>(x, WV, Vt);

    const size_t smemBytes = 32768 + 131072 + 65536;   // 224 KB of 320 KB WGP LDS
    attn<<<dim3(SEQ / 16, BATCH), 256, smemBytes, stream>>>(Qw, Kw, Vt, out);
}
